// GraphRegression_41618233098730
// MI455X (gfx1250) — matrix operand order repaired and checked by reading, not compiled, not run on hardware
//
#include <hip/hip_runtime.h>
#include <hip/hip_bf16.h>

typedef __attribute__((ext_vector_type(16))) _Float16 v16h;
typedef __attribute__((ext_vector_type(8)))  _Float16 v8h;
typedef __attribute__((ext_vector_type(8)))  float    v8f;

#define BN 64      // nodes
#define BD 256     // feature dim
#define DH 128     // projection dim

// LDS pitches (in halves / floats) chosen for 16-byte fragment alignment
// and conflict-free 16-lane row access (row stride not a multiple of 64 dwords).
#define XH_P  264   // xh / aggh pitch (halves)
#define XT_P  72    // xhT pitch (halves)
#define QK_P  136   // qh / kh pitch (halves)
#define SC_P  68    // scores pitch (floats)
#define AH_P  72    // A_hat pitch (halves)

__device__ inline v8f vzero8() {
    v8f z;
#pragma unroll
    for (int i = 0; i < 8; ++i) z[i] = 0.0f;
    return z;
}

__device__ inline float wave_sum(float v) {
#pragma unroll
    for (int off = 16; off; off >>= 1) v += __shfl_xor(v, off, 32);
    return v;
}
__device__ inline float wave_max(float v) {
#pragma unroll
    for (int off = 16; off; off >>= 1) v = fmaxf(v, __shfl_xor(v, off, 32));
    return v;
}

// A operand (16x32 f16, M = rows): lane L holds A[m0 + L%16][c0..c0+7] and [c0+16..c0+23],
// c0 = klo + 8*(L/16).  (ISA 7.12.2, 16-bit A-matrix table)
__device__ inline v16h load_frag_A(const _Float16* base, int pitch, int m0, int klo, int lane) {
    int r  = m0 + (lane & 15);
    int c0 = klo + 8 * (lane >> 4);
    union { v16h v; v8h h[2]; } u;
    u.h[0] = *(const v8h*)(base + r * pitch + c0);
    u.h[1] = *(const v8h*)(base + r * pitch + c0 + 16);
    return u.v;
}

// B operand (32x16 f16): lane L holds Bt[n0 + L%16][k0..k0+7] and [k0+16..k0+23], k0 = klo + 8*(L/16),
// where Bt is B transposed stored row-major with given pitch. (ISA 7.12.2 B layout)
__device__ inline v16h load_frag_B(const _Float16* base, int pitch, int n0, int klo, int lane) {
    int c  = n0 + (lane & 15);
    int k0 = klo + 8 * (lane >> 4);
    union { v16h v; v8h h[2]; } u;
    u.h[0] = *(const v8h*)(base + c * pitch + k0);
    u.h[1] = *(const v8h*)(base + c * pitch + k0 + 16);
    return u.v;
}

__device__ inline v8f wmma_f16(v16h a, v16h b, v8f c) {
    return __builtin_amdgcn_wmma_f32_16x16x32_f16(false, a, false, b, (short)0, c, false, false);
}

// ---------------------------------------------------------------------------
// fp32 -> f16 weight conversion
__global__ void cvt_f16_kernel(const float* __restrict__ s, _Float16* __restrict__ d, int n) {
    int i = blockIdx.x * blockDim.x + threadIdx.x;
    if (i < n) d[i] = (_Float16)s[i];
}

// row softmax of a 64x64 matrix (static adjacency), fp32 out
__global__ void softmax64_kernel(const float* __restrict__ adj, float* __restrict__ out) {
    int w = threadIdx.x >> 5, lane = threadIdx.x & 31;
#pragma unroll
    for (int i = 0; i < 8; ++i) {
        int r = w * 8 + i;
        float v0 = adj[r * 64 + lane];
        float v1 = adj[r * 64 + 32 + lane];
        float mx = wave_max(fmaxf(v0, v1));
        float e0 = __expf(v0 - mx), e1 = __expf(v1 - mx);
        float inv = 1.0f / wave_sum(e0 + e1);
        out[r * 64 + lane]      = e0 * inv;
        out[r * 64 + 32 + lane] = e1 * inv;
    }
}

// ---------------------------------------------------------------------------
// One full graph-conv layer (LN[+leaky] -> q/k proj -> scores -> softmax+Astat
// -> aggregate -> FC), one workgroup (8 waves) per batch element.
__global__ void __launch_bounds__(256, 1)
layer_kernel(const float* __restrict__ xin, float* __restrict__ out,
             const float* __restrict__ ln_g, const float* __restrict__ ln_b,
             const float* __restrict__ Astat,
             const _Float16* __restrict__ thh, const _Float16* __restrict__ phh,
             const _Float16* __restrict__ fcwh, const float* __restrict__ fcb,
             int leaky)
{
    __shared__ __align__(16) _Float16 xh [BN * XH_P];   // LN(x) f16, row-major; reused as aggh
    __shared__ __align__(16) _Float16 xhT[BD * XT_P];   // LN(x) f16, transposed
    __shared__ __align__(16) _Float16 qh [BN * QK_P];   // q (pre-scaled by 1/sqrt(DH))
    __shared__ __align__(16) _Float16 kh [BN * QK_P];   // k
    __shared__ __align__(16) float    sc [BN * SC_P];   // raw scores f32
    __shared__ __align__(16) _Float16 Ah [BN * AH_P];   // A_static + softmax(scores), f16

    const int w = threadIdx.x >> 5, lane = threadIdx.x & 31;
    const size_t slab = (size_t)blockIdx.x * BN * BD;

    // ---- Stage L: LayerNorm (+ optional LeakyReLU) -> xh (f16) and xhT ----
    {
        const float4* gp = (const float4*)ln_g;
        const float4* bp = (const float4*)ln_b;
        float4 g0 = gp[lane], g1 = gp[lane + 32];
        float4 b0 = bp[lane], b1 = bp[lane + 32];
        float gg[8] = {g0.x, g0.y, g0.z, g0.w, g1.x, g1.y, g1.z, g1.w};
        float bb[8] = {b0.x, b0.y, b0.z, b0.w, b1.x, b1.y, b1.z, b1.w};
#pragma unroll
        for (int i = 0; i < 8; ++i) {
            int r = w * 8 + i;
            const float4* row = (const float4*)(xin + slab + (size_t)r * BD);
            float4 a = row[lane], c = row[lane + 32];
            float v[8] = {a.x, a.y, a.z, a.w, c.x, c.y, c.z, c.w};
            float s = 0.f, ss = 0.f;
#pragma unroll
            for (int d = 0; d < 8; ++d) { s += v[d]; ss += v[d] * v[d]; }
            s = wave_sum(s); ss = wave_sum(ss);
            float mean = s * (1.0f / 256.0f);
            float inv  = rsqrtf(ss * (1.0f / 256.0f) - mean * mean + 1e-5f);
#pragma unroll
            for (int d = 0; d < 8; ++d) {
                float y = (v[d] - mean) * inv * gg[d] + bb[d];
                if (leaky) y = (y >= 0.f) ? y : 0.1f * y;
                int col = (d < 4) ? (lane * 4 + d) : (128 + lane * 4 + d - 4);
                _Float16 hv = (_Float16)y;
                xh [r * XH_P + col] = hv;
                xhT[col * XT_P + r] = hv;
            }
        }
    }
    __syncthreads();

    // ---- Stage P: q = x @ th^T (scaled), k = x @ ph^T.  M=64,N=128,K=256 ----
    // q and k share every A fragment: load A once, issue two WMMAs.
    {
        const float qscale = 0.08838834764831845f;   // 1/sqrt(128)
        const int n0 = w * 16;                        // each wave owns one N-tile of DH
        v8f accq[4] = {vzero8(), vzero8(), vzero8(), vzero8()};
        v8f acck[4] = {vzero8(), vzero8(), vzero8(), vzero8()};
        for (int kk = 0; kk < 8; ++kk) {
            int klo = kk * 32;
            v16h bq = load_frag_B(thh, BD, n0, klo, lane);   // global, L2-resident
            v16h bk = load_frag_B(phh, BD, n0, klo, lane);
            v16h af[4];
#pragma unroll
            for (int m = 0; m < 4; ++m) af[m] = load_frag_A(xh, XH_P, m * 16, klo, lane);
#pragma unroll
            for (int m = 0; m < 4; ++m) {
                accq[m] = wmma_f16(af[m], bq, accq[m]);
                acck[m] = wmma_f16(af[m], bk, acck[m]);
            }
        }
#pragma unroll
        for (int m = 0; m < 4; ++m)
#pragma unroll
            for (int r = 0; r < 8; ++r) {
                int row = (m * 16 + (lane >> 4) * 8 + r) * QK_P + n0 + (lane & 15);
                qh[row] = (_Float16)(accq[m][r] * qscale);
                kh[row] = (_Float16)(acck[m][r]);
            }
    }
    __syncthreads();

    // ---- Stage S: scores = q @ k^T.  M=64,N=64,K=128 ----
    // wave w owns tiles (m = w&3, n0) and (m, n0+32): shared A fragment.
    {
        int m  = (w & 3) * 16;
        int n0 = (w >> 2) * 16;
        v8f acc0 = vzero8(), acc1 = vzero8();
        for (int kk = 0; kk < 4; ++kk) {
            int klo = kk * 32;
            v16h af = load_frag_A(qh, QK_P, m, klo, lane);
            v16h b0 = load_frag_B(kh, QK_P, n0,      klo, lane);
            v16h b1 = load_frag_B(kh, QK_P, n0 + 32, klo, lane);
            acc0 = wmma_f16(af, b0, acc0);
            acc1 = wmma_f16(af, b1, acc1);
        }
#pragma unroll
        for (int r = 0; r < 8; ++r) {
            int row = (m + (lane >> 4) * 8 + r) * SC_P + (lane & 15);
            sc[row + n0]      = acc0[r];
            sc[row + n0 + 32] = acc1[r];
        }
    }
    __syncthreads();

    // ---- Stage SM: A_hat = A_static + softmax(scores) -> f16 ----
#pragma unroll
    for (int i = 0; i < 8; ++i) {
        int r = w * 8 + i;
        float v0 = sc[r * SC_P + lane], v1 = sc[r * SC_P + 32 + lane];
        float mx = wave_max(fmaxf(v0, v1));
        float e0 = __expf(v0 - mx), e1 = __expf(v1 - mx);
        float inv = 1.0f / wave_sum(e0 + e1);
        Ah[r * AH_P + lane]      = (_Float16)(e0 * inv + Astat[r * 64 + lane]);
        Ah[r * AH_P + 32 + lane] = (_Float16)(e1 * inv + Astat[r * 64 + 32 + lane]);
    }
    __syncthreads();

    _Float16* aggh = xh;   // xh is dead after Stage P: reuse as agg buffer

    // ---- Stage AGG: agg = A_hat @ x.  M=64,N=256,K=64 (B operand from xhT) ----
    // wave w owns N-tiles w*16 and (w+8)*16 for all 4 M-tiles: shared A fragments.
    {
        int n0 = w * 16, n1 = (w + 8) * 16;
        v8f acc0[4] = {vzero8(), vzero8(), vzero8(), vzero8()};
        v8f acc1[4] = {vzero8(), vzero8(), vzero8(), vzero8()};
#pragma unroll
        for (int kk = 0; kk < 2; ++kk) {
            int klo = kk * 32;
            v16h b0 = load_frag_B(xhT, XT_P, n0, klo, lane);
            v16h b1 = load_frag_B(xhT, XT_P, n1, klo, lane);
            v16h af[4];
#pragma unroll
            for (int m = 0; m < 4; ++m) af[m] = load_frag_A(Ah, AH_P, m * 16, klo, lane);
#pragma unroll
            for (int m = 0; m < 4; ++m) {
                acc0[m] = wmma_f16(af[m], b0, acc0[m]);
                acc1[m] = wmma_f16(af[m], b1, acc1[m]);
            }
        }
#pragma unroll
        for (int m = 0; m < 4; ++m)
#pragma unroll
            for (int r = 0; r < 8; ++r) {
                int row = (m * 16 + (lane >> 4) * 8 + r) * XH_P + (lane & 15);
                aggh[row + n0] = (_Float16)acc0[m][r];
                aggh[row + n1] = (_Float16)acc1[m][r];
            }
    }
    __syncthreads();

    // ---- Stage FC: out = agg @ fcw^T + fcb.  M=64,K=256,N=256 ----
    // wave w owns N-tiles w*16 and (w+8)*16: shared A fragments.
    {
        int n0 = w * 16, n1 = (w + 8) * 16;
        v8f acc0[4] = {vzero8(), vzero8(), vzero8(), vzero8()};
        v8f acc1[4] = {vzero8(), vzero8(), vzero8(), vzero8()};
        for (int kk = 0; kk < 8; ++kk) {
            int klo = kk * 32;
            v16h b0 = load_frag_B(fcwh, BD, n0, klo, lane);   // fcw [D,D] row-major == Bt
            v16h b1 = load_frag_B(fcwh, BD, n1, klo, lane);
            v16h af[4];
#pragma unroll
            for (int m = 0; m < 4; ++m) af[m] = load_frag_A(aggh, XH_P, m * 16, klo, lane);
#pragma unroll
            for (int m = 0; m < 4; ++m) {
                acc0[m] = wmma_f16(af[m], b0, acc0[m]);
                acc1[m] = wmma_f16(af[m], b1, acc1[m]);
            }
        }
        int c0 = n0 + (lane & 15), c1 = n1 + (lane & 15);
        float bias0 = fcb[c0], bias1 = fcb[c1];
        float* ob = out + slab;
#pragma unroll
        for (int m = 0; m < 4; ++m)
#pragma unroll
            for (int r = 0; r < 8; ++r) {
                size_t row = (size_t)(m * 16 + (lane >> 4) * 8 + r) * BD;
                ob[row + c0] = acc0[m][r] + bias0;
                ob[row + c1] = acc1[m][r] + bias1;
            }
    }
}

// ---------------------------------------------------------------------------
extern "C" void kernel_launch(void* const* d_in, const int* in_sizes, int n_in,
                              void* d_out, int out_size, void* d_ws, size_t ws_size,
                              hipStream_t stream) {
    const float* graph   = (const float*)d_in[0];
    const float* ln_in_g = (const float*)d_in[1];
    const float* ln_in_b = (const float*)d_in[2];
    const float* adj0    = (const float*)d_in[3];
    const float* th0     = (const float*)d_in[4];
    const float* ph0     = (const float*)d_in[5];
    const float* fcw0    = (const float*)d_in[6];
    const float* fcb0    = (const float*)d_in[7];
    const float* ln0_g   = (const float*)d_in[8];
    const float* ln0_b   = (const float*)d_in[9];
    const float* adj1    = (const float*)d_in[10];
    const float* th1     = (const float*)d_in[11];
    const float* ph1     = (const float*)d_in[12];
    const float* fcw1    = (const float*)d_in[13];
    const float* fcb1    = (const float*)d_in[14];
    float* out = (float*)d_out;

    // workspace: f16 weights (L2-resident) + static adjacency softmax
    _Float16* th0h  = (_Float16*)d_ws;
    _Float16* ph0h  = th0h  + DH * BD;
    _Float16* fcw0h = ph0h  + DH * BD;
    _Float16* th1h  = fcw0h + BD * BD;
    _Float16* ph1h  = th1h  + DH * BD;
    _Float16* fcw1h = ph1h  + DH * BD;
    float*    As0   = (float*)(fcw1h + BD * BD);
    float*    As1   = As0 + BN * BN;

    cvt_f16_kernel<<<(DH * BD + 255) / 256, 256, 0, stream>>>(th0,  th0h,  DH * BD);
    cvt_f16_kernel<<<(DH * BD + 255) / 256, 256, 0, stream>>>(ph0,  ph0h,  DH * BD);
    cvt_f16_kernel<<<(BD * BD + 255) / 256, 256, 0, stream>>>(fcw0, fcw0h, BD * BD);
    cvt_f16_kernel<<<(DH * BD + 255) / 256, 256, 0, stream>>>(th1,  th1h,  DH * BD);
    cvt_f16_kernel<<<(DH * BD + 255) / 256, 256, 0, stream>>>(ph1,  ph1h,  DH * BD);
    cvt_f16_kernel<<<(BD * BD + 255) / 256, 256, 0, stream>>>(fcw1, fcw1h, BD * BD);
    softmax64_kernel<<<1, 256, 0, stream>>>(adj0, As0);
    softmax64_kernel<<<1, 256, 0, stream>>>(adj1, As1);

    const int B = 2048;
    // layer 0: graph -> d_out
    layer_kernel<<<B, 256, 0, stream>>>(graph, out, ln_in_g, ln_in_b, As0,
                                        th0h, ph0h, fcw0h, fcb0, 0);
    // layer 1: in-place on d_out (each block reads only its own slab fully
    // into LDS before writing it back)
    layer_kernel<<<B, 256, 0, stream>>>(out, out, ln0_g, ln0_b, As1,
                                        th1h, ph1h, fcw1h, fcb1, 1);
}
